// StateQuantizer_4879082848595
// MI455X (gfx1250) — matrix operand order repaired and checked by reading, not compiled, not run on hardware
//
#include <hip/hip_runtime.h>
#include <stdint.h>

// Problem constants (fixed by the reference)
#define B_ROWS 8192
#define C_DIM  2048            // N*D = 32*64
#define K_CODES 1024
#define NCHUNK (C_DIM / 32)    // 64 K-chunks of width 32
#define FRAG_ELEMS 512         // 32 lanes * 16 bf16 per fragment

typedef __attribute__((ext_vector_type(16))) __bf16 v16bf;
typedef __attribute__((ext_vector_type(8)))  __bf16 v8bf;
typedef __attribute__((ext_vector_type(8)))  float  v8f;

// ---------- helpers ----------

// Monotone order-preserving key for float -> uint32 (handles negatives)
__device__ __forceinline__ uint32_t fkey(float f) {
  uint32_t u = __float_as_uint(f);
  return (u & 0x80000000u) ? ~u : (u | 0x80000000u);
}
__device__ __forceinline__ float funkey(uint32_t u) {
  uint32_t b = (u & 0x80000000u) ? (u ^ 0x80000000u) : ~u;
  return __uint_as_float(b);
}

__device__ __forceinline__ v8f wmma_bf16(v16bf a, v16bf b, v8f c) {
  return __builtin_amdgcn_wmma_f32_16x16x32_bf16(
      /*neg_a=*/false, a, /*neg_b=*/false, b,
      /*c_mod=*/(short)0, c, /*reuse_a=*/false, /*reuse_b=*/false);
}

// Split 16 fp32 into bf16 hi + lo (fallback path, in-register)
__device__ __forceinline__ void split16(float4 a0, float4 a1, float4 a2, float4 a3,
                                        v16bf& hi, v16bf& lo) {
  float f[16] = {a0.x, a0.y, a0.z, a0.w, a1.x, a1.y, a1.z, a1.w,
                 a2.x, a2.y, a2.z, a2.w, a3.x, a3.y, a3.z, a3.w};
#pragma unroll
  for (int i = 0; i < 16; ++i) {
    __bf16 h = (__bf16)f[i];
    hi[i] = h;
    lo[i] = (__bf16)(f[i] - (float)h);
  }
}

// Fragment-layout destination offset (in elements) for 8 contiguous values of
// logical (row, col): row tile r0 = row/16, chunk c0 = col/32.
// A-fragment lane map: lane = half*16 + row%16, half 0 holds K [0..8)+[16..24),
// half 1 holds K [8..16)+[24..32) (ISA 7.12.2, 16-bit A 16x32).
// B-fragment lane map: lane = half*16 + code%16, same K split as the A-fragment.
__device__ __forceinline__ size_t frag_off_A(int row, int col) {
  int r0 = row >> 4, l16 = row & 15;
  int c0 = col >> 5, w = col & 31;
  int half = (w >> 3) & 1;          // 0,8,16,24 -> 0,1,0,1
  int elembase = (w >> 4) << 3;     // 0,8
  return ((size_t)(r0 * NCHUNK + c0) * 32 + half * 16 + l16) * 16 + elembase;
}
__device__ __forceinline__ size_t frag_off_B(int code, int col) {
  int n0 = code >> 4, l16 = code & 15;
  int c0 = col >> 5, w = col & 31;
  int half = (w >> 3) & 1;          // 0,8,16,24 -> 0,1,0,1
  int elembase = (w >> 4) << 3;     // 0,8
  return ((size_t)(n0 * NCHUNK + c0) * 32 + half * 16 + l16) * 16 + elembase;
}

// ---------- kernel: init packed (dist,index) mins ----------
__global__ void k_init(unsigned long long* __restrict__ packed) {
  int i = blockIdx.x * blockDim.x + threadIdx.x;
  if (i < B_ROWS) packed[i] = 0xFFFFFFFFFFFFFFFFull;
}

// ---------- kernel: z -> bf16 hi/lo in A-fragment layout + z^2 partials ------
__global__ __launch_bounds__(256) void k_prep_z(const float* __restrict__ z,
                                                __bf16* __restrict__ zhi,
                                                __bf16* __restrict__ zlo,
                                                float* __restrict__ part) {
  __shared__ float red[256];
  const size_t n8 = (size_t)B_ROWS * C_DIM / 8;
  float s = 0.f;
  for (size_t i = (size_t)blockIdx.x * 256 + threadIdx.x; i < n8;
       i += (size_t)gridDim.x * 256) {
    const size_t i8 = i * 8;
    const int row = (int)(i8 >> 11);       // / C_DIM
    const int col = (int)(i8 & (C_DIM - 1));
    const float4* p = (const float4*)(z + i8);
    float4 x0 = p[0], x1 = p[1];
    float f[8] = {x0.x, x0.y, x0.z, x0.w, x1.x, x1.y, x1.z, x1.w};
    v8bf h, l;
#pragma unroll
    for (int j = 0; j < 8; ++j) {
      __bf16 hb = (__bf16)f[j];
      h[j] = hb;
      l[j] = (__bf16)(f[j] - (float)hb);
      s += f[j] * f[j];
    }
    const size_t o = frag_off_A(row, col);
    *(v8bf*)(zhi + o) = h;
    *(v8bf*)(zlo + o) = l;
  }
  red[threadIdx.x] = s;
  __syncthreads();
  for (int off = 128; off > 0; off >>= 1) {
    if (threadIdx.x < off) red[threadIdx.x] += red[threadIdx.x + off];
    __syncthreads();
  }
  if (threadIdx.x == 0) part[blockIdx.x] = red[0];
}

// ---------- kernel: E -> bf16 hi/lo in B-fragment layout + e_norm ------------
__global__ __launch_bounds__(256) void k_prep_e(const float* __restrict__ e,
                                                __bf16* __restrict__ ehi,
                                                __bf16* __restrict__ elo,
                                                float* __restrict__ enorm) {
  __shared__ float red[256];
  const int k = blockIdx.x;
  const float* row = e + (size_t)k * C_DIM;
  float s = 0.f;
  for (int i = threadIdx.x; i < C_DIM / 8; i += 256) {
    const int col = i * 8;
    const float4* p = (const float4*)(row + col);
    float4 x0 = p[0], x1 = p[1];
    float f[8] = {x0.x, x0.y, x0.z, x0.w, x1.x, x1.y, x1.z, x1.w};
    v8bf h, l;
#pragma unroll
    for (int j = 0; j < 8; ++j) {
      __bf16 hb = (__bf16)f[j];
      h[j] = hb;
      l[j] = (__bf16)(f[j] - (float)hb);
      s += f[j] * f[j];
    }
    const size_t o = frag_off_B(k, col);
    *(v8bf*)(ehi + o) = h;
    *(v8bf*)(elo + o) = l;
  }
  red[threadIdx.x] = s;
  __syncthreads();
  for (int off = 128; off > 0; off >>= 1) {
    if (threadIdx.x < off) red[threadIdx.x] += red[threadIdx.x + off];
    __syncthreads();
  }
  if (threadIdx.x == 0) enorm[k] = red[0];
}

// ---------- kernel: plain z^2 partials (fallback path) ----------
__global__ void k_zsq(const float* __restrict__ z, float* __restrict__ part) {
  __shared__ float red[256];
  const float4* z4 = (const float4*)z;
  const size_t n4 = (size_t)B_ROWS * C_DIM / 4;
  float s = 0.f;
  for (size_t i = (size_t)blockIdx.x * 256 + threadIdx.x; i < n4; i += 256 * 256) {
    float4 v = z4[i];
    s += v.x * v.x + v.y * v.y + v.z * v.z + v.w * v.w;
  }
  red[threadIdx.x] = s;
  __syncthreads();
  for (int off = 128; off > 0; off >>= 1) {
    if (threadIdx.x < off) red[threadIdx.x] += red[threadIdx.x + off];
    __syncthreads();
  }
  if (threadIdx.x == 0) part[blockIdx.x] = red[0];
}

// ---------- kernel: e_norm only (fallback path) ----------
__global__ void k_enorm(const float* __restrict__ e, float* __restrict__ enorm) {
  __shared__ float red[256];
  int k = blockIdx.x;
  const float4* row = (const float4*)(e + (size_t)k * C_DIM);
  float s = 0.f;
  for (int i = threadIdx.x; i < C_DIM / 4; i += 256) {
    float4 v = row[i];
    s += v.x * v.x + v.y * v.y + v.z * v.z + v.w * v.w;
  }
  red[threadIdx.x] = s;
  __syncthreads();
  for (int off = 128; off > 0; off >>= 1) {
    if (threadIdx.x < off) red[threadIdx.x] += red[threadIdx.x + off];
    __syncthreads();
  }
  if (threadIdx.x == 0) enorm[k] = red[0];
}

// ---------- epilogue shared by both GEMM variants ----------
// C/D layout: VGPR j, lanes 0-15 -> row j, lanes 16-31 -> row j+8; col = l16.
__device__ __forceinline__ void argmin_epilogue(
    const v8f acc[4], const float en[4], int colbase0, int rowbase, int half,
    int l16, unsigned long long* __restrict__ packed) {
#pragma unroll
  for (int j = 0; j < 8; ++j) {
    float bd = en[0] - 2.f * acc[0][j];
    int   bi = colbase0 + l16;
#pragma unroll
    for (int t = 1; t < 4; ++t) {
      float d = en[t] - 2.f * acc[t][j];
      int   i = colbase0 + t * 16 + l16;
      if (d < bd) { bd = d; bi = i; }  // ascending index: ties keep earlier
    }
#pragma unroll
    for (int m = 1; m < 16; m <<= 1) {  // reduce within 16-lane half
      float od = __shfl_xor(bd, m, 32);
      int   oi = __shfl_xor(bi, m, 32);
      if (od < bd || (od == bd && oi < bi)) { bd = od; bi = oi; }
    }
    if (l16 == 0) {
      unsigned long long pk = ((unsigned long long)fkey(bd) << 32) | (unsigned)bi;
      atomicMin(&packed[rowbase + j + half * 8], pk);
    }
  }
}

// ---------- kernel: WMMA distance GEMM + argmin, fragment-layout operands ----
// Block tile 128 rows x 128 codes, 512 threads = 16 waves.
// All operand loads are lane-contiguous 32 B (fully coalesced 1 KB per wave).
__global__ __launch_bounds__(512) void k_distmin_bf(
    const __bf16* __restrict__ zhi, const __bf16* __restrict__ zlo,
    const __bf16* __restrict__ ehi, const __bf16* __restrict__ elo,
    const float* __restrict__ enorm, unsigned long long* __restrict__ packed) {
  const int lane = threadIdx.x & 31;
  const int wave = threadIdx.x >> 5;
  const int half = lane >> 4;
  const int l16  = lane & 15;
  const int rowbase  = blockIdx.x * 128 + (wave & 7) * 16;
  const int colbase0 = blockIdx.y * 128 + (wave >> 3) * 64;

  // Fragment bases (elements): advance by FRAG_ELEMS per K-chunk.
  const int r0 = rowbase >> 4;
  const int n0 = colbase0 >> 4;
  size_t aoff = ((size_t)r0 * NCHUNK * 32 + lane) * 16;
  size_t boff = ((size_t)n0 * NCHUNK * 32 + lane) * 16;
  const size_t btile = (size_t)NCHUNK * FRAG_ELEMS;  // next 16-code tile

  v8f acc[4] = {};

  for (int c0 = 0; c0 < NCHUNK; ++c0) {
    v16bf ahi = *(const v16bf*)(zhi + aoff);
    v16bf alo = *(const v16bf*)(zlo + aoff);
#pragma unroll
    for (int t = 0; t < 4; ++t) {
      const size_t bo = boff + (size_t)t * btile;
      v16bf bhi = *(const v16bf*)(ehi + bo);
      v16bf blo = *(const v16bf*)(elo + bo);
      acc[t] = wmma_bf16(ahi, bhi, acc[t]);
      acc[t] = wmma_bf16(ahi, blo, acc[t]);
      acc[t] = wmma_bf16(alo, bhi, acc[t]);
    }
    aoff += FRAG_ELEMS;
    boff += FRAG_ELEMS;
  }

  float en[4];
#pragma unroll
  for (int t = 0; t < 4; ++t) en[t] = enorm[colbase0 + t * 16 + l16];
  argmin_epilogue(acc, en, colbase0, rowbase, half, l16, packed);
}

// ---------- kernel: fallback GEMM (fp32 source, convert in-register) ----------
__global__ __launch_bounds__(512) void k_distmin_f32(
    const float* __restrict__ z, const float* __restrict__ e,
    const float* __restrict__ enorm, unsigned long long* __restrict__ packed) {
  const int lane = threadIdx.x & 31;
  const int wave = threadIdx.x >> 5;
  const int half = lane >> 4;
  const int l16  = lane & 15;
  const int rowbase  = blockIdx.x * 128 + (wave & 7) * 16;
  const int colbase0 = blockIdx.y * 128 + (wave >> 3) * 64;

  const float* zrow = z + (size_t)(rowbase + l16) * C_DIM;
  v8f acc[4] = {};

  for (int cc = 0; cc < C_DIM; cc += 32) {
    const int ka = cc + half * 8;
    float4 a0 = *(const float4*)(zrow + ka);
    float4 a1 = *(const float4*)(zrow + ka + 4);
    float4 a2 = *(const float4*)(zrow + ka + 16);
    float4 a3 = *(const float4*)(zrow + ka + 20);
    v16bf ahi, alo;
    split16(a0, a1, a2, a3, ahi, alo);

#pragma unroll
    for (int t = 0; t < 4; ++t) {
      const float* erow =
          e + (size_t)(colbase0 + t * 16 + l16) * C_DIM + cc + half * 8;
      float4 b0 = *(const float4*)(erow);
      float4 b1 = *(const float4*)(erow + 4);
      float4 b2 = *(const float4*)(erow + 16);
      float4 b3 = *(const float4*)(erow + 20);
      v16bf bhi, blo;
      split16(b0, b1, b2, b3, bhi, blo);
      acc[t] = wmma_bf16(ahi, bhi, acc[t]);
      acc[t] = wmma_bf16(ahi, blo, acc[t]);
      acc[t] = wmma_bf16(alo, bhi, acc[t]);
    }
  }

  float en[4];
#pragma unroll
  for (int t = 0; t < 4; ++t) en[t] = enorm[colbase0 + t * 16 + l16];
  argmin_epilogue(acc, en, colbase0, rowbase, half, l16, packed);
}

// ---------- kernel: gather z_q[b] = E[ind[b]] ----------
__global__ void k_gather(const float* __restrict__ e,
                         const unsigned long long* __restrict__ packed,
                         float* __restrict__ out) {
  int b = blockIdx.x;
  unsigned code = (unsigned)(packed[b] & 0xFFFFFFFFull);
  const float4* src = (const float4*)(e + (size_t)code * C_DIM);
  float4* dst = (float4*)(out + (size_t)b * C_DIM);
  for (int i = threadIdx.x; i < C_DIM / 4; i += blockDim.x) dst[i] = src[i];
}

// ---------- kernel: loss = 12.5 * (sum z^2 + sum min(e_norm-2S)) / (B*C) ------
__global__ void k_loss(const float* __restrict__ part, int npart,
                       const unsigned long long* __restrict__ packed,
                       float* __restrict__ out_loss) {
  __shared__ float red[256];
  float s = 0.f;
  for (int i = threadIdx.x; i < npart; i += 256) s += part[i];
  for (int b = threadIdx.x; b < B_ROWS; b += 256)
    s += funkey((uint32_t)(packed[b] >> 32));
  red[threadIdx.x] = s;
  __syncthreads();
  for (int off = 128; off > 0; off >>= 1) {
    if (threadIdx.x < off) red[threadIdx.x] += red[threadIdx.x + off];
    __syncthreads();
  }
  if (threadIdx.x == 0)
    out_loss[0] = 12.5f * red[0] / (float)((size_t)B_ROWS * C_DIM);
}

// ---------- launch ----------
extern "C" void kernel_launch(void* const* d_in, const int* in_sizes, int n_in,
                              void* d_out, int out_size, void* d_ws, size_t ws_size,
                              hipStream_t stream) {
  const float* z = (const float*)d_in[0];  // [8192, 32, 64] fp32
  const float* e = (const float*)d_in[1];  // [1024, 2048]  fp32
  float* out = (float*)d_out;              // [8192*2048] z_q + [1] loss

  char* ws = (char*)d_ws;
  unsigned long long* packed = (unsigned long long*)ws;   // 8192 * 8B
  float* enorm = (float*)(ws + 65536);                    // 1024 * 4B
  float* part  = (float*)(ws + 65536 + 4096);             // up to 1024 * 4B
  char*  big   = ws + 65536 + 4096 + 4096;                // bf16 staging

  const size_t z_elems = (size_t)B_ROWS * C_DIM;   // 16,777,216
  const size_t e_elems = (size_t)K_CODES * C_DIM;  //  2,097,152
  const size_t need = 65536 + 4096 + 4096 + 2 * 2 * z_elems + 2 * 2 * e_elems;

  k_init<<<(B_ROWS + 255) / 256, 256, 0, stream>>>(packed);
  dim3 grid(B_ROWS / 128, K_CODES / 128);  // (64, 8)

  if (ws_size >= need) {
    // Fast path: pre-convert once into WMMA-fragment-ordered bf16 hi/lo
    // (fused with the norm passes); GEMM loop is coalesced loads + WMMA only.
    __bf16* zhi = (__bf16*)big;
    __bf16* zlo = zhi + z_elems;
    __bf16* ehi = zlo + z_elems;
    __bf16* elo = ehi + e_elems;
    k_prep_z<<<1024, 256, 0, stream>>>(z, zhi, zlo, part);
    k_prep_e<<<K_CODES, 256, 0, stream>>>(e, ehi, elo, enorm);
    k_distmin_bf<<<grid, 512, 0, stream>>>(zhi, zlo, ehi, elo, enorm, packed);
    k_gather<<<B_ROWS, 256, 0, stream>>>(e, packed, out);
    k_loss<<<1, 256, 0, stream>>>(part, 1024, packed, out + z_elems);
  } else {
    // Fallback: convert inside the GEMM (small workspace).
    k_enorm<<<K_CODES, 256, 0, stream>>>(e, enorm);
    k_zsq<<<256, 256, 0, stream>>>(z, part);
    k_distmin_f32<<<grid, 512, 0, stream>>>(z, e, enorm, packed);
    k_gather<<<B_ROWS, 256, 0, stream>>>(e, packed, out);
    k_loss<<<1, 256, 0, stream>>>(part, 256, packed, out + z_elems);
  }
}
